// CausalSelfAttention_21646635172351
// MI455X (gfx1250) — compile-verified, matrix operand order checked by reading, not run on hardware
//
#include <hip/hip_runtime.h>
#include <hip/hip_bf16.h>
#include <math.h>

// ---------------------------------------------------------------------------
// CDNA5 (gfx1250) causal self-attention block, bf16 WMMA pipeline.
//   B=4, T=2048, D=1024, H=16, HD=64
// GEMMs use double-buffered GLOBAL_LOAD_ASYNC_TO_LDS staging when available.
// ---------------------------------------------------------------------------

typedef __bf16 bf16;
typedef __attribute__((ext_vector_type(16))) __bf16 v16bf;
typedef __attribute__((ext_vector_type(8)))  __bf16 v8bf;
typedef __attribute__((ext_vector_type(8)))  float  v8f;
typedef int v4i_gcc __attribute__((vector_size(16)));

#if defined(__has_builtin)
#if __has_builtin(__builtin_amdgcn_global_load_async_to_lds_b128)
#define ATTN_HAS_ASYNC_LDS 1
#endif
#endif

// Copy 16 bytes (8 bf16) per lane from global to LDS.
// Async (ASYNCcnt-tracked, no VGPR round-trip) when the builtin exists.
__device__ __forceinline__ void cp16_to_lds(bf16* ldst, const bf16* gsrc) {
#if defined(ATTN_HAS_ASYNC_LDS)
  typedef __attribute__((address_space(1))) v4i_gcc* gp_t;  // global int4*
  typedef __attribute__((address_space(3))) v4i_gcc* lp_t;  // LDS int4*
  __builtin_amdgcn_global_load_async_to_lds_b128(
      (gp_t)(uintptr_t)gsrc,
      (lp_t)(unsigned)(uintptr_t)ldst,   // low 32 bits = LDS byte offset
      /*imm offset=*/0, /*cpol=*/0);
#else
  *(v8bf*)ldst = *(const v8bf*)gsrc;
#endif
}

__device__ __forceinline__ void async_wait0() {
#if defined(ATTN_HAS_ASYNC_LDS) && __has_builtin(__builtin_amdgcn_s_wait_asynccnt)
  __builtin_amdgcn_s_wait_asynccnt(0);
#endif
}

__device__ __forceinline__ v8f wmma_bf16(v16bf a, v16bf b, v8f c) {
  // D = A(16x32) * B(32x16) + C, f32 accumulate
  return __builtin_amdgcn_wmma_f32_16x16x32_bf16(
      /*neg_a=*/false, a, /*neg_b=*/false, b,
      /*c_mod=*/(short)0, c, /*reuse_a=*/false, /*reuse_b=*/false);
}

// Load one 16-bit A/B WMMA fragment from a row-major [rows x ld] bf16 matrix.
// Per CDNA5 ISA layout: lane L holds row (caller passes row = base + L%16),
// elements K = 8*(L>>4) + 0..7 (vgpr0-3) and 16 + 8*(L>>4) + 0..7 (vgpr4-7).
__device__ __forceinline__ v16bf frag_load(const bf16* __restrict__ base,
                                           int row, int ld, int k0) {
  const int half = (threadIdx.x & 31) >> 4;
  const bf16* p = base + (size_t)row * (size_t)ld + k0 + (half << 3);
  union { v16bf v; v8bf h[2]; } u;
  u.h[0] = *(const v8bf*)(p);
  u.h[1] = *(const v8bf*)(p + 16);
  return u.v;
}

// ---------------------------------------------------------------------------
// fp32 -> bf16 cast
// ---------------------------------------------------------------------------
__global__ void cvt_f32_to_bf16(const float* __restrict__ s,
                                bf16* __restrict__ d, size_t n) {
  size_t i = (size_t)blockIdx.x * blockDim.x + threadIdx.x;
  size_t stride = (size_t)gridDim.x * blockDim.x;
  for (; i < n; i += stride) d[i] = (bf16)s[i];
}

// ---------------------------------------------------------------------------
// Tiled WMMA GEMM:  C[M x N] = A[M x K] (bf16, row-major) * W[N x K]^T (bf16)
// Block tile 128x128, 8 waves (256 thr), each wave 32(M) x 64(N), K-step 32.
// LDS double-buffered with async global->LDS staging.
// mode 0: store fp32 into outF[M x N]
// mode 1: qkv scatter epilogue -> q/k [B,H,T,HD] bf16, vT [B,H,HD,T] bf16
// ---------------------------------------------------------------------------
__global__ __launch_bounds__(256)
void gemm_wmma(const bf16* __restrict__ A, const bf16* __restrict__ W,
               int M, int N, int K,
               float* __restrict__ outF,
               bf16* __restrict__ qO, bf16* __restrict__ kO,
               bf16* __restrict__ vTO, int mode) {
  __shared__ __align__(16) bf16 As[2][128 * 32];
  __shared__ __align__(16) bf16 Bs[2][128 * 32];

  const int tid   = threadIdx.x;
  const int lane  = tid & 31;
  const int col16 = lane & 15;
  const int half  = lane >> 4;
  const int wave  = tid >> 5;   // 0..7
  const int wm    = wave >> 1;  // 0..3 -> M offset wm*32
  const int wn    = wave & 1;   // 0..1 -> N offset wn*64
  const int bm    = blockIdx.x * 128;
  const int bn    = blockIdx.y * 128;

  v8f acc[2][4];
#pragma unroll
  for (int i = 0; i < 2; ++i)
#pragma unroll
    for (int j = 0; j < 4; ++j)
#pragma unroll
      for (int r = 0; r < 8; ++r) acc[i][j][r] = 0.0f;

  // Per-thread staging slot: row ldr (0..127), 16-element segment seg.
  const int ldr = tid >> 1;
  const int seg = (tid & 1) << 4;
  const bf16* gA = A + (size_t)(bm + ldr) * K + seg;
  const bf16* gW = W + (size_t)(bn + ldr) * K + seg;
  bf16* sA[2] = { &As[0][ldr * 32 + seg], &As[1][ldr * 32 + seg] };
  bf16* sB[2] = { &Bs[0][ldr * 32 + seg], &Bs[1][ldr * 32 + seg] };

  // Prologue: stage K-slab 0 into buffer 0.
  cp16_to_lds(sA[0],     gA);
  cp16_to_lds(sA[0] + 8, gA + 8);
  cp16_to_lds(sB[0],     gW);
  cp16_to_lds(sB[0] + 8, gW + 8);
  async_wait0();
  __syncthreads();

  int cur = 0;
  for (int k0 = 0; k0 < K; k0 += 32) {
    const int nxt = cur ^ 1;
    if (k0 + 32 < K) {  // stage next slab while computing this one
      cp16_to_lds(sA[nxt],     gA + k0 + 32);
      cp16_to_lds(sA[nxt] + 8, gA + k0 + 40);
      cp16_to_lds(sB[nxt],     gW + k0 + 32);
      cp16_to_lds(sB[nxt] + 8, gW + k0 + 40);
    }

    const bf16* aB = &As[cur][0];
    const bf16* bB = &Bs[cur][0];
    v16bf a0 = frag_load(aB, wm * 32 + col16,      32, 0);
    v16bf a1 = frag_load(aB, wm * 32 + 16 + col16, 32, 0);
    v16bf b0 = frag_load(bB, wn * 64 +  0 + col16, 32, 0);
    v16bf b1 = frag_load(bB, wn * 64 + 16 + col16, 32, 0);
    v16bf b2 = frag_load(bB, wn * 64 + 32 + col16, 32, 0);
    v16bf b3 = frag_load(bB, wn * 64 + 48 + col16, 32, 0);

    acc[0][0] = wmma_bf16(a0, b0, acc[0][0]);
    acc[0][1] = wmma_bf16(a0, b1, acc[0][1]);
    acc[0][2] = wmma_bf16(a0, b2, acc[0][2]);
    acc[0][3] = wmma_bf16(a0, b3, acc[0][3]);
    acc[1][0] = wmma_bf16(a1, b0, acc[1][0]);
    acc[1][1] = wmma_bf16(a1, b1, acc[1][1]);
    acc[1][2] = wmma_bf16(a1, b2, acc[1][2]);
    acc[1][3] = wmma_bf16(a1, b3, acc[1][3]);

    async_wait0();
    __syncthreads();
    cur = nxt;
  }

  // Epilogue. C layout: element r of v8f -> row (r + 8*half), col = lane&15.
#pragma unroll
  for (int mi = 0; mi < 2; ++mi) {
#pragma unroll
    for (int nj = 0; nj < 4; ++nj) {
      const int gmB = bm + wm * 32 + mi * 16;
      const int gnB = bn + wn * 64 + nj * 16;
#pragma unroll
      for (int r = 0; r < 8; ++r) {
        const int row = gmB + r + 8 * half;
        const int col = gnB + col16;
        const float v = acc[mi][nj][r];
        if (mode == 0) {
          outF[(size_t)row * N + col] = v;
        } else {
          // row -> (b, t); col -> {q|k|v}, head h, head-dim hd
          const int b  = row >> 11;        // T = 2048
          const int t  = row & 2047;
          const int sc = col >> 10;        // D = 1024 per section
          const int cc = col & 1023;
          const int h  = cc >> 6;          // HD = 64
          const int hd = cc & 63;
          const size_t bh = (size_t)b * 16 + h;
          const bf16 bv = (bf16)v;
          if (sc == 0)      qO[(bh * 2048 + t) * 64 + hd] = bv;
          else if (sc == 1) kO[(bh * 2048 + t) * 64 + hd] = bv;
          else              vTO[(bh * 64 + hd) * 2048 + t] = bv;  // transposed
        }
      }
    }
  }
}

// ---------------------------------------------------------------------------
// Flash attention: one wave per (b, h, 16-query tile).
// S = (Q K^T) * 1/8 with causal mask, online softmax, O = P V via Vᵀ.
// ---------------------------------------------------------------------------
__global__ __launch_bounds__(32)
void attn_wmma(const bf16* __restrict__ Q, const bf16* __restrict__ Kt,
               const bf16* __restrict__ Vt, bf16* __restrict__ Y) {
  const int T = 2048, HD = 64, H = 16, D = 1024;
  const int tile = blockIdx.x;
  const int q0   = (tile % (T / 16)) * 16;
  const int bh   = tile / (T / 16);
  const int h    = bh % H;
  const int b    = bh / H;

  const bf16* qb = Q  + (size_t)bh * T * HD;
  const bf16* kb = Kt + (size_t)bh * T * HD;
  const bf16* vb = Vt + (size_t)bh * HD * T;   // [HD][T]

  const int lane  = threadIdx.x & 31;
  const int col16 = lane & 15;
  const int half  = lane >> 4;

  __shared__ __align__(16) bf16 P[16 * 32];

  // Q A-fragments for the two 32-wide K(=head-dim) steps, loaded once.
  const v16bf aq0 = frag_load(qb, q0 + col16, HD, 0);
  const v16bf aq1 = frag_load(qb, q0 + col16, HD, 32);

  v8f of[4];
  float m[8], l[8];
#pragma unroll
  for (int j = 0; j < 4; ++j)
#pragma unroll
    for (int r = 0; r < 8; ++r) of[j][r] = 0.0f;
#pragma unroll
  for (int r = 0; r < 8; ++r) { m[r] = -INFINITY; l[r] = 0.0f; }

  const float scale = 0.125f;  // 1/sqrt(64)
  const int tEnd = q0 + 16;

  for (int t0 = 0; t0 < tEnd; t0 += 32) {
    if (t0 + 32 < tEnd) {  // prefetch next key chunk into near caches
      __builtin_prefetch(kb + (size_t)(t0 + 32 + col16) * HD, 0, 3);
      __builtin_prefetch(kb + (size_t)(t0 + 48 + col16) * HD, 0, 3);
      __builtin_prefetch(vb + (size_t)col16 * T + t0 + 32, 0, 3);
    }

    // ---- S = Q K^T over a 16(q) x 32(k) chunk: two 16x16 C tiles ----
    v8f s0, s1;
#pragma unroll
    for (int r = 0; r < 8; ++r) { s0[r] = 0.0f; s1[r] = 0.0f; }
    {
      v16bf b00 = frag_load(kb, t0 +      col16, HD, 0);
      v16bf b01 = frag_load(kb, t0 +      col16, HD, 32);
      v16bf b10 = frag_load(kb, t0 + 16 + col16, HD, 0);
      v16bf b11 = frag_load(kb, t0 + 16 + col16, HD, 32);
      s0 = wmma_bf16(aq0, b00, s0);
      s0 = wmma_bf16(aq1, b01, s0);
      s1 = wmma_bf16(aq0, b10, s1);
      s1 = wmma_bf16(aq1, b11, s1);
    }

    // ---- online softmax update (rows r+8*half, cols across 16 lanes) ----
    float p0[8], p1[8], corr[8];
#pragma unroll
    for (int r = 0; r < 8; ++r) {
      const int qrow = q0 + r + 8 * half;
      const int c0   = t0 + col16;
      const int c1   = c0 + 16;
      float x0 = (c0 <= qrow) ? s0[r] * scale : -INFINITY;
      float x1 = (c1 <= qrow) ? s1[r] * scale : -INFINITY;
      float mx = fmaxf(x0, x1);
      mx = fmaxf(mx, __shfl_xor(mx, 1, 32));
      mx = fmaxf(mx, __shfl_xor(mx, 2, 32));
      mx = fmaxf(mx, __shfl_xor(mx, 4, 32));
      mx = fmaxf(mx, __shfl_xor(mx, 8, 32));
      const float mn = fmaxf(m[r], mx);
      corr[r] = __expf(m[r] - mn);
      const float e0 = __expf(x0 - mn);
      const float e1 = __expf(x1 - mn);
      float rs = e0 + e1;
      rs += __shfl_xor(rs, 1, 32);
      rs += __shfl_xor(rs, 2, 32);
      rs += __shfl_xor(rs, 4, 32);
      rs += __shfl_xor(rs, 8, 32);
      l[r] = l[r] * corr[r] + rs;
      m[r] = mn;
      p0[r] = e0;
      p1[r] = e1;
    }
#pragma unroll
    for (int j = 0; j < 4; ++j)
#pragma unroll
      for (int r = 0; r < 8; ++r) of[j][r] *= corr[r];

    // ---- C-layout -> A-fragment layout via LDS ----
#pragma unroll
    for (int r = 0; r < 8; ++r) {
      P[(r + 8 * half) * 32 + col16]      = (bf16)p0[r];
      P[(r + 8 * half) * 32 + col16 + 16] = (bf16)p1[r];
    }
    __syncthreads();  // single-wave WG: ordering fence for LDS
    const v16bf pA = frag_load(P, col16, 32, 0);

    // ---- O += P * V  (B from Vᵀ rows: contiguous) ----
    v16bf bv0 = frag_load(vb,  0 + col16, T, t0);
    v16bf bv1 = frag_load(vb, 16 + col16, T, t0);
    v16bf bv2 = frag_load(vb, 32 + col16, T, t0);
    v16bf bv3 = frag_load(vb, 48 + col16, T, t0);
    of[0] = wmma_bf16(pA, bv0, of[0]);
    of[1] = wmma_bf16(pA, bv1, of[1]);
    of[2] = wmma_bf16(pA, bv2, of[2]);
    of[3] = wmma_bf16(pA, bv3, of[3]);
    __syncthreads();
  }

  // ---- finalize: O / l, write bf16 y[B*T, D] at col h*64 + d ----
  float invl[8];
#pragma unroll
  for (int r = 0; r < 8; ++r) invl[r] = 1.0f / l[r];
#pragma unroll
  for (int j = 0; j < 4; ++j)
#pragma unroll
    for (int r = 0; r < 8; ++r) {
      const int t   = q0 + r + 8 * half;
      const int col = h * 64 + j * 16 + col16;
      Y[((size_t)b * T + t) * D + col] = (bf16)(of[j][r] * invl[r]);
    }
}

// ---------------------------------------------------------------------------
// Host-side orchestration
// ---------------------------------------------------------------------------
extern "C" void kernel_launch(void* const* d_in, const int* in_sizes, int n_in,
                              void* d_out, int out_size, void* d_ws, size_t ws_size,
                              hipStream_t stream) {
  const float* x      = (const float*)d_in[0];
  const float* w_qkv  = (const float*)d_in[1];
  const float* w_proj = (const float*)d_in[2];
  float* out = (float*)d_out;

  const int B = 4, T = 2048, D = 1024, H = 16;
  const size_t M  = (size_t)B * T;        // 8192
  const size_t MD = M * D;                // 8M elements

  // workspace layout (bf16): x | w_qkv | w_proj | q | k | vT | y  (~88 MB)
  bf16* x_bf  = (bf16*)d_ws;
  bf16* wq_bf = x_bf  + MD;
  bf16* wp_bf = wq_bf + (size_t)3 * D * D;
  bf16* qb    = wp_bf + (size_t)D * D;
  bf16* kb    = qb + MD;
  bf16* vTb   = kb + MD;
  bf16* yb    = vTb + MD;

  // 1) casts
  cvt_f32_to_bf16<<<2048, 256, 0, stream>>>(x, x_bf, MD);
  cvt_f32_to_bf16<<<1024, 256, 0, stream>>>(w_qkv, wq_bf, (size_t)3 * D * D);
  cvt_f32_to_bf16<<<512, 256, 0, stream>>>(w_proj, wp_bf, (size_t)D * D);

  // 2) QKV GEMM (M=8192, N=3072, K=1024) with fused Q/K/Vᵀ scatter
  {
    dim3 grid((unsigned)(M / 128), (unsigned)((3 * D) / 128));
    gemm_wmma<<<grid, 256, 0, stream>>>(x_bf, wq_bf, (int)M, 3 * D, D,
                                        nullptr, qb, kb, vTb, /*mode=*/1);
  }

  // 3) flash attention: one wave per (b, h, 16-query tile)
  attn_wmma<<<B * H * (T / 16), 32, 0, stream>>>(qb, kb, vTb, yb);

  // 4) output projection GEMM (M=8192, N=1024, K=1024) -> fp32 out
  {
    dim3 grid((unsigned)(M / 128), (unsigned)(D / 128));
    gemm_wmma<<<grid, 256, 0, stream>>>(yb, wp_bf, (int)M, D, D,
                                        out, nullptr, nullptr, nullptr, /*mode=*/0);
  }
}
